// SoftMoE5_3719441678640
// MI455X (gfx1250) — matrix operand order repaired and checked by reading, not compiled, not run on hardware
//
#include <hip/hip_runtime.h>

typedef __attribute__((ext_vector_type(16))) __bf16 v16bf;
typedef __attribute__((ext_vector_type(8)))  __bf16 v8bf;
typedef __attribute__((ext_vector_type(8)))  float  v8f;
typedef __attribute__((ext_vector_type(4)))  unsigned int v4u;

#define B_  64
#define N_  256
#define D_  1024
#define E_  16
#define ES_ 256
#define H_  256

// ---------------- helpers ----------------

static __device__ __forceinline__ unsigned short f2bf(float f) {
  unsigned u = __float_as_uint(f);
  u += 0x7fffu + ((u >> 16) & 1u);     // round-to-nearest-even
  return (unsigned short)(u >> 16);
}

static __device__ __forceinline__ float block_sum256(float v) {
  __shared__ float sh[256];
  int t = threadIdx.x;
  sh[t] = v; __syncthreads();
  for (int s = 128; s > 0; s >>= 1) {
    if (t < s) sh[t] += sh[t + s];
    __syncthreads();
  }
  float r = sh[0]; __syncthreads();
  return r;
}

static __device__ __forceinline__ float block_max256(float v) {
  __shared__ float sh[256];
  int t = threadIdx.x;
  sh[t] = v; __syncthreads();
  for (int s = 128; s > 0; s >>= 1) {
    if (t < s) sh[t] = fmaxf(sh[t], sh[t + s]);
    __syncthreads();
  }
  float r = sh[0]; __syncthreads();
  return r;
}

// Load a 16-element bf16 fragment as two contiguous 16B chunks (p, p+delta)
static __device__ __forceinline__ v16bf ld_frag(const unsigned short* p, int delta) {
  union U { v16bf f; v8bf h[2]; } u;
  u.h[0] = *(const v8bf*)(p);
  u.h[1] = *(const v8bf*)(p + delta);
  return u.f;
}

#define WMMA_BF16(a, b, c) \
  __builtin_amdgcn_wmma_f32_16x16x32_bf16(false, (a), false, (b), (short)0, (c), false, false)

// One wave computes a 16(M) x 64(N) strip of C = A[16xK] * B[Kx64].
// Simple fully-unrolled loop: loads are issued as clauses right before the
// WMMAs; no manual pipelining (register pressure must stay below the spill
// threshold — 1 fragment set + 32 accumulator VGPRs). Latency is hidden by
// the 8 waves per block across the WGP's SIMD32s.
static __device__ __forceinline__ void wmma_tile(
    const unsigned short* __restrict__ Arow,
    const unsigned short* __restrict__ B0,
    const unsigned short* __restrict__ B1,
    const unsigned short* __restrict__ B2,
    const unsigned short* __restrict__ B3,
    int K, int half, v8f c[4])
{
  const int ao = half * 16;   // A: element e -> K = half*16 + e
  const int bo = half * 16;   // B: element e -> K = half*16 + e
  for (int k = 0; k < K; k += 32) {
    v16bf a  = ld_frag(Arow + k + ao, 8);
    v16bf b0 = ld_frag(B0 + k + bo, 8);
    v16bf b1 = ld_frag(B1 + k + bo, 8);
    v16bf b2 = ld_frag(B2 + k + bo, 8);
    v16bf b3 = ld_frag(B3 + k + bo, 8);
    c[0] = WMMA_BF16(a, b0, c[0]);
    c[1] = WMMA_BF16(a, b1, c[1]);
    c[2] = WMMA_BF16(a, b2, c[2]);
    c[3] = WMMA_BF16(a, b3, c[3]);
  }
}

// ---------------- preprocessing kernels ----------------

// LayerNorm + L2-normalize each token; write x_unit*scale (bf16, [B,N,D])
// and xn transposed (bf16, [B,D,N]) for the slot_in GEMM B operand.
__global__ void k_ln(const float* __restrict__ x, const float* __restrict__ g,
                     const float* __restrict__ be, const float* __restrict__ scl,
                     unsigned short* __restrict__ xu, unsigned short* __restrict__ xnT)
{
  int token = blockIdx.x;            // b*N + n
  int b = token >> 8;                // N_ = 256
  int n = token & 255;
  const float* xp = x + (size_t)token * D_;
  int t = threadIdx.x;

  float v[4]; float s = 0.f, s2 = 0.f;
  #pragma unroll
  for (int i = 0; i < 4; ++i) { v[i] = xp[t * 4 + i]; s += v[i]; s2 += v[i] * v[i]; }
  s  = block_sum256(s);
  s2 = block_sum256(s2);
  float mean = s * (1.0f / D_);
  float var  = s2 * (1.0f / D_) - mean * mean;
  float rstd = rsqrtf(var + 1e-5f);

  float xn[4]; float l2 = 0.f;
  #pragma unroll
  for (int i = 0; i < 4; ++i) {
    int d = t * 4 + i;
    xn[i] = (v[i] - mean) * rstd * g[d] + be[d];
    l2 += xn[i] * xn[i];
  }
  l2 = block_sum256(l2);
  float inv = scl[0] / fmaxf(sqrtf(l2), 1e-12f);

  #pragma unroll
  for (int i = 0; i < 4; ++i) {
    int d = t * 4 + i;
    xu[(size_t)token * D_ + d]          = f2bf(xn[i] * inv);
    xnT[((size_t)b * D_ + d) * N_ + n]  = f2bf(xn[i]);
  }
}

// L2-normalize mu over D per (e,s) column; write transposed bf16 [ES,D].
__global__ void k_mu(const float* __restrict__ mu, unsigned short* __restrict__ mu_t)
{
  int es = blockIdx.x;
  int t = threadIdx.x;
  float v[4]; float s2 = 0.f;
  #pragma unroll
  for (int i = 0; i < 4; ++i) {
    int d = t * 4 + i;
    v[i] = mu[(size_t)d * ES_ + es];
    s2 += v[i] * v[i];
  }
  s2 = block_sum256(s2);
  float inv = 1.0f / fmaxf(sqrtf(s2), 1e-12f);
  #pragma unroll
  for (int i = 0; i < 4; ++i) {
    int d = t * 4 + i;
    mu_t[(size_t)es * D_ + d] = f2bf(v[i] * inv);
  }
}

// w1 [E,D,H] -> w1t bf16 [E,H,D]
__global__ void k_cvt_w1(const float* __restrict__ w1, unsigned short* __restrict__ w1t)
{
  size_t idx = (size_t)blockIdx.x * 256 + threadIdx.x;    // < E*D*H
  int e  = (int)(idx / (D_ * H_));
  int r  = (int)(idx % (D_ * H_));
  int d  = r / H_;
  int hc = r % H_;
  w1t[((size_t)e * H_ + hc) * D_ + d] = f2bf(w1[idx]);
}

// w2 [E,H,D] -> w2t bf16 [E,D,H]
__global__ void k_cvt_w2(const float* __restrict__ w2, unsigned short* __restrict__ w2t)
{
  size_t idx = (size_t)blockIdx.x * 256 + threadIdx.x;    // < E*H*D
  int e  = (int)(idx / (H_ * D_));
  int r  = (int)(idx % (H_ * D_));
  int hc = r / D_;
  int d  = r % D_;
  w2t[((size_t)e * D_ + d) * H_ + hc] = f2bf(w2[idx]);
}

// ---------------- GEMM kernels ----------------

// logits[b] [N,ES] = xu[b] [N,D] x mu_unit [D,ES]   (Bt = mu_t [ES,D])
__global__ void k_gemm_logits(const unsigned short* __restrict__ xu,
                              const unsigned short* __restrict__ mu_t,
                              float* __restrict__ lg)
{
  int wid = threadIdx.x >> 5, lane = threadIdx.x & 31;
  int half = lane >> 4, nl = lane & 15;
  int b  = blockIdx.x >> 3;
  int gw = ((blockIdx.x & 7) << 3) | wid;    // 0..63
  int m0 = (gw >> 2) * 16;
  int n0 = (gw & 3) * 64;

  const unsigned short* A  = xu + (size_t)b * N_ * D_ + (size_t)(m0 + nl) * D_;
  const unsigned short* B0 = mu_t + (size_t)(n0 + nl) * D_;
  v8f c[4] = {};
  wmma_tile(A, B0, B0 + 16 * D_, B0 + 32 * D_, B0 + 48 * D_, D_, half, c);

  float* Cp = lg + (size_t)b * N_ * ES_;
  #pragma unroll
  for (int j = 0; j < 4; ++j)
    #pragma unroll
    for (int r = 0; r < 8; ++r)
      Cp[(size_t)(m0 + half * 8 + r) * ES_ + (n0 + j * 16 + nl)] = c[j][r];
}

// dispatch = softmax over tokens n (per b,es); store transposed bf16 [B,ES,N]
__global__ void k_softmax_n(const float* __restrict__ lg, unsigned short* __restrict__ dpT)
{
  int b  = blockIdx.x >> 8;
  int es = blockIdx.x & 255;
  int n  = threadIdx.x;
  float v = lg[((size_t)b * N_ + n) * ES_ + es];
  float m = block_max256(v);
  float e = expf(v - m);
  float s = block_sum256(e);
  dpT[((size_t)b * ES_ + es) * N_ + n] = f2bf(e / s);
}

// combine = softmax over flattened (e,s) per token; bf16 [B,N,ES]
__global__ void k_softmax_es(const float* __restrict__ lg, unsigned short* __restrict__ cmb)
{
  int b  = blockIdx.x >> 8;
  int n  = blockIdx.x & 255;
  int es = threadIdx.x;
  float v = lg[((size_t)b * N_ + n) * ES_ + es];
  float m = block_max256(v);
  float e = expf(v - m);
  float s = block_sum256(e);
  cmb[((size_t)b * N_ + n) * ES_ + es] = f2bf(e / s);
}

// slot_in[b] [ES,D] = dispatch^T [ES,N] x xn [N,D]   (Bt = xnT [D,N])
__global__ void k_gemm_slotin(const unsigned short* __restrict__ dpT,
                              const unsigned short* __restrict__ xnT,
                              unsigned short* __restrict__ sin_)
{
  int wid = threadIdx.x >> 5, lane = threadIdx.x & 31;
  int half = lane >> 4, nl = lane & 15;
  int b  = blockIdx.x >> 5;
  int gw = ((blockIdx.x & 31) << 3) | wid;   // 0..255
  int m0 = (gw >> 4) * 16;
  int n0 = (gw & 15) * 64;

  const unsigned short* A  = dpT + (size_t)b * ES_ * N_ + (size_t)(m0 + nl) * N_;
  const unsigned short* B0 = xnT + (size_t)b * D_ * N_ + (size_t)(n0 + nl) * N_;
  v8f c[4] = {};
  wmma_tile(A, B0, B0 + 16 * N_, B0 + 32 * N_, B0 + 48 * N_, N_, half, c);

  unsigned short* Cp = sin_ + (size_t)b * ES_ * D_;
  #pragma unroll
  for (int j = 0; j < 4; ++j)
    #pragma unroll
    for (int r = 0; r < 8; ++r)
      Cp[(size_t)(m0 + half * 8 + r) * D_ + (n0 + j * 16 + nl)] = f2bf(c[j][r]);
}

// h[b,e] [16,H] = gelu(slot_in[b,e] [16,D] x w1[e] [D,H] + b1[e])   (Bt = w1t[e] [H,D])
__global__ void k_gemm_mlp1(const unsigned short* __restrict__ sin_,
                            const unsigned short* __restrict__ w1t,
                            const float* __restrict__ b1,
                            unsigned short* __restrict__ hbuf)
{
  int wid = threadIdx.x >> 5, lane = threadIdx.x & 31;
  int half = lane >> 4, nl = lane & 15;
  int gwave = blockIdx.x * 8 + wid;          // 0..4095
  int pair  = gwave >> 2;                    // b*E + e
  int nt    = gwave & 3;
  int b = pair >> 4, e = pair & 15;
  int n0 = nt * 64;

  const unsigned short* A  = sin_ + ((size_t)b * ES_ + e * 16 + nl) * D_;
  const unsigned short* B0 = w1t + (size_t)e * H_ * D_ + (size_t)(n0 + nl) * D_;
  v8f c[4] = {};
  wmma_tile(A, B0, B0 + 16 * D_, B0 + 32 * D_, B0 + 48 * D_, D_, half, c);

  #pragma unroll
  for (int j = 0; j < 4; ++j) {
    int n = n0 + j * 16 + nl;
    float bias = b1[e * H_ + n];
    #pragma unroll
    for (int r = 0; r < 8; ++r) {
      int m = half * 8 + r;
      float v = c[j][r] + bias;
      v = 0.5f * v * (1.0f + erff(v * 0.70710678118f));   // exact GELU
      hbuf[((size_t)b * ES_ + e * 16 + m) * H_ + n] = f2bf(v);
    }
  }
}

// slot_out[b,e] [16,D] = h[b,e] [16,H] x w2[e] [H,D] + b2[e]  (Bt = w2t[e] [D,H])
// stored TRANSPOSED bf16 [B,D,ES] -> contiguous 16B store per lane.
__global__ void k_gemm_mlp2(const unsigned short* __restrict__ hbuf,
                            const unsigned short* __restrict__ w2t,
                            const float* __restrict__ b2,
                            unsigned short* __restrict__ soT)
{
  int wid = threadIdx.x >> 5, lane = threadIdx.x & 31;
  int half = lane >> 4, nl = lane & 15;
  int gwave = blockIdx.x * 8 + wid;          // 0..16383
  int pair  = gwave >> 4;                    // b*E + e
  int nt    = gwave & 15;
  int b = pair >> 4, e = pair & 15;
  int n0 = nt * 64;

  const unsigned short* A  = hbuf + ((size_t)b * ES_ + e * 16 + nl) * H_;
  const unsigned short* B0 = w2t + (size_t)e * D_ * H_ + (size_t)(n0 + nl) * H_;
  v8f c[4] = {};
  wmma_tile(A, B0, B0 + 16 * H_, B0 + 32 * H_, B0 + 48 * H_, H_, half, c);

  #pragma unroll
  for (int j = 0; j < 4; ++j) {
    int n = n0 + j * 16 + nl;                // global d column
    float bias = b2[e * D_ + n];
    unsigned int pk[4];
    #pragma unroll
    for (int q = 0; q < 4; ++q) {
      unsigned int lo = f2bf(c[j][2 * q] + bias);
      unsigned int hi = f2bf(c[j][2 * q + 1] + bias);
      pk[q] = lo | (hi << 16);
    }
    v4u* dst = (v4u*)(soT + ((size_t)b * D_ + n) * ES_ + e * 16 + half * 8);
    *dst = (v4u){pk[0], pk[1], pk[2], pk[3]};
  }
}

// out[b] [N,D] = combine [N,ES] x slot_out [ES,D]   (Bt = soT [D,ES]), fp32 out
__global__ void k_gemm_out(const unsigned short* __restrict__ cmb,
                           const unsigned short* __restrict__ soT,
                           float* __restrict__ out)
{
  int wid = threadIdx.x >> 5, lane = threadIdx.x & 31;
  int half = lane >> 4, nl = lane & 15;
  int b  = blockIdx.x >> 5;
  int gw = ((blockIdx.x & 31) << 3) | wid;   // 0..255
  int m0 = (gw >> 4) * 16;
  int n0 = (gw & 15) * 64;

  const unsigned short* A  = cmb + (size_t)b * N_ * ES_ + (size_t)(m0 + nl) * ES_;
  const unsigned short* B0 = soT + (size_t)b * D_ * ES_ + (size_t)(n0 + nl) * ES_;
  v8f c[4] = {};
  wmma_tile(A, B0, B0 + 16 * ES_, B0 + 32 * ES_, B0 + 48 * ES_, ES_, half, c);

  float* Cp = out + (size_t)b * N_ * D_;
  #pragma unroll
  for (int j = 0; j < 4; ++j)
    #pragma unroll
    for (int r = 0; r < 8; ++r)
      Cp[(size_t)(m0 + half * 8 + r) * D_ + (n0 + j * 16 + nl)] = c[j][r];
}

// ---------------- launch ----------------

extern "C" void kernel_launch(void* const* d_in, const int* in_sizes, int n_in,
                              void* d_out, int out_size, void* d_ws, size_t ws_size,
                              hipStream_t stream) {
  const float* x   = (const float*)d_in[0];
  const float* gam = (const float*)d_in[1];
  const float* bet = (const float*)d_in[2];
  const float* mu  = (const float*)d_in[3];
  const float* scl = (const float*)d_in[4];
  const float* w1  = (const float*)d_in[5];
  const float* b1  = (const float*)d_in[6];
  const float* w2  = (const float*)d_in[7];
  const float* b2  = (const float*)d_in[8];
  float* out = (float*)d_out;

  char* ws = (char*)d_ws;
  const size_t MB = 1ull << 20;
  unsigned short* xu   = (unsigned short*)(ws + 0);         // [B,N,D]  bf16  32MB
  unsigned short* xnT  = (unsigned short*)(ws + 32 * MB);   // [B,D,N]  bf16  32MB
  unsigned short* mu_t = (unsigned short*)(ws + 64 * MB);   // [ES,D]   bf16  0.5MB
  float*          lg   = (float*)        (ws + 65 * MB);    // [B,N,ES] f32   16MB
  unsigned short* dpT  = (unsigned short*)(ws + 81 * MB);   // [B,ES,N] bf16   8MB
  unsigned short* cmb  = (unsigned short*)(ws + 89 * MB);   // [B,N,ES] bf16   8MB
  unsigned short* sin_ = (unsigned short*)(ws + 97 * MB);   // [B,ES,D] bf16  32MB
  unsigned short* hbuf = (unsigned short*)(ws + 129 * MB);  // [B,ES,H] bf16   8MB
  unsigned short* soT  = (unsigned short*)(ws + 137 * MB);  // [B,D,ES] bf16  32MB
  unsigned short* w1t  = (unsigned short*)(ws + 169 * MB);  // [E,H,D]  bf16   8MB
  unsigned short* w2t  = (unsigned short*)(ws + 177 * MB);  // [E,D,H]  bf16   8MB

  k_ln<<<B_ * N_, 256, 0, stream>>>(x, gam, bet, scl, xu, xnT);
  k_mu<<<ES_, 256, 0, stream>>>(mu, mu_t);
  k_cvt_w1<<<(E_ * D_ * H_) / 256, 256, 0, stream>>>(w1, w1t);
  k_cvt_w2<<<(E_ * H_ * D_) / 256, 256, 0, stream>>>(w2, w2t);

  k_gemm_logits<<<B_ * 8, 256, 0, stream>>>(xu, mu_t, lg);
  k_softmax_n<<<B_ * ES_, 256, 0, stream>>>(lg, dpT);
  k_softmax_es<<<B_ * N_, 256, 0, stream>>>(lg, cmb);

  k_gemm_slotin<<<B_ * 32, 256, 0, stream>>>(dpT, xnT, sin_);
  k_gemm_mlp1<<<(B_ * E_ * 4) / 8, 256, 0, stream>>>(sin_, w1t, b1, hbuf);
  k_gemm_mlp2<<<(B_ * E_ * 16) / 8, 256, 0, stream>>>(hbuf, w2t, b2, soT);
  k_gemm_out<<<B_ * 32, 256, 0, stream>>>(cmb, soT, out);
}
